// VectQuantize_30829275251358
// MI455X (gfx1250) — matrix operand order repaired and checked by reading, not compiled, not run on hardware
//
#include <hip/hip_runtime.h>

typedef unsigned short u16;
typedef __attribute__((ext_vector_type(16))) __bf16 bf16x16;
typedef __attribute__((ext_vector_type(8)))  float  f32x8;

union Frag {
  bf16x16 v;
  uint4 q[2];
};

__device__ __forceinline__ u16 f2bf(float f) {
  unsigned u = __float_as_uint(f);
  u += 0x7FFFu + ((u >> 16) & 1u);   // round-to-nearest-even
  return (u16)(u >> 16);
}
__device__ __forceinline__ float bf2f(u16 h) {
  return __uint_as_float(((unsigned)h) << 16);
}

// CDNA5 async global->LDS copy (ASYNCcnt path). The LDS destination byte
// offset is the low 32 bits of the flat shared address (LDS aperture lives in
// the high bits), per the ISA flat-address mapping.
__device__ __forceinline__ void async_copy_b128(void* lds, const void* gmem) {
  unsigned loff = (unsigned)(unsigned long long)lds;
  asm volatile("global_load_async_to_lds_b128 %0, %1, off"
               :
               : "v"(loff), "v"(gmem)
               : "memory");
}
__device__ __forceinline__ void wait_async0() {
  asm volatile("s_wait_asynccnt 0x0" ::: "memory");
}

#define WMMA_BF16 __builtin_amdgcn_wmma_f32_16x16x32_bf16

// ---------------------------------------------------------------------------
// Kernel 0: codebook fp32 -> bf16 hi/lo split + per-code squared norms.
// ---------------------------------------------------------------------------
__global__ void vq_prep(const float* __restrict__ w, u16* __restrict__ wh,
                        u16* __restrict__ wl, float* __restrict__ wsq) {
  int k = blockIdx.x, t = threadIdx.x;          // k: code 0..1023, t: channel
  float v = w[k * 256 + t];
  u16 h = f2bf(v);
  wh[k * 256 + t] = h;
  wl[k * 256 + t] = f2bf(v - bf2f(h));
  __shared__ float s[256];
  s[t] = v * v;
  __syncthreads();
  for (int o = 128; o > 0; o >>= 1) {
    if (t < o) s[t] += s[t + o];
    __syncthreads();
  }
  if (t == 0) wsq[k] = s[0];
}

// ---------------------------------------------------------------------------
// Kernel 1: per-WG: 64 pixels, all 1024 codes. 4 waves * 16 pixels.
//  dist surrogate s_k = |w_k|^2 - 2 x.w_k   (|x|^2 constant per row).
//  x.w via split-bf16: xh*wh + xl*wh + xh*wl, fp32 WMMA accumulate.
//  B tiles double-buffered in LDS via global_load_async_to_lds_b128.
// ---------------------------------------------------------------------------
__global__ __launch_bounds__(128) void vq_main(
    const float* __restrict__ x, const float* __restrict__ w,
    const u16* __restrict__ wh, const u16* __restrict__ wl,
    const float* __restrict__ wsq, float* __restrict__ out,
    float* __restrict__ partial) {
  constexpr int PIX = 64, D = 256;
  constexpr int SSTRIDE = 264;      // ushort row stride (16B aligned, padded)
  constexpr int FSTRIDE = 261;      // float row stride for phase-2 staging
  constexpr int BROW = 264;         // B tile row stride (ushorts), padded
  constexpr int BTILE = 16 * BROW;  // one 16-code tile (hi or lo), ushorts

  extern __shared__ __align__(16) char smem[];
  u16* xh = (u16*)smem;                 // phase 1: 64*264*2 = 33792 B
  u16* xl = xh + PIX * SSTRIDE;         //          33792 B (total 67584 B)
  u16* bbuf = (u16*)smem;               // ct loop: 2 x (Bh+Bl) = 2*2*8448 B
  float* wst = (float*)smem;            // phase 2: 64*261*4 = 66816 B
  __shared__ float wsq_lds[1024];
  __shared__ int idx_lds[PIX];
  __shared__ float red[128];

  const int tid = threadIdx.x;
  const int lane = tid & 31;
  const int wv = tid >> 5;       // wave 0..3 -> pixel tile
  const int m = lane & 15;       // A row / B column within 16x16 tile
  const int hh = lane >> 4;      // half-wave select for K interleave
  const int wg = blockIdx.x;
  const int b = (wg * PIX) >> 10;       // batch (H*W = 1024)
  const int hw0 = (wg * PIX) & 1023;    // 64 consecutive hw positions
  const size_t xbase = (size_t)b * 262144 + (size_t)hw0;

  // ---- Phase 1: stage x tile (NCHW, contiguous along hw) into LDS bf16 hi/lo
  for (int i = tid; i < PIX * D; i += 128) {
    int ch = i >> 6, p = i & 63;                       // coalesced 256B runs
    float v = x[xbase + (size_t)ch * 1024 + p];
    u16 h = f2bf(v);
    xh[p * SSTRIDE + ch] = h;
    xl[p * SSTRIDE + ch] = f2bf(v - bf2f(h));
  }
  for (int i = tid; i < 1024; i += 128) wsq_lds[i] = wsq[i];
  __syncthreads();

  // ---- A fragments in registers (pixels as M, channels as K).
  // 16-bit A 16x32 layout: low lanes hold K=k0+0..7 (e0..7), k0+16..23
  // (e8..15); high lanes hold +8 within each 16-block.
  bf16x16 Ah[8], Al[8];
  {
    const int arow = (wv * 16 + m) * SSTRIDE;
    #pragma unroll
    for (int kc = 0; kc < 8; ++kc) {
      const int off = arow + kc * 32 + hh * 8;
      Frag a, al;
      a.q[0]  = *(const uint4*)(xh + off);
      a.q[1]  = *(const uint4*)(xh + off + 16);
      al.q[0] = *(const uint4*)(xl + off);
      al.q[1] = *(const uint4*)(xl + off + 16);
      Ah[kc] = a.v;
      Al[kc] = al.v;
    }
  }
  __syncthreads();   // x tile dead; smem region now reused for B buffers

  // Async-stage one 16-code B tile (hi+lo) into buffer `buf`.
  // 512 16B-chunks per array; 128 threads x 4 iters; 8 async instrs per wave.
  auto stage_tile = [&](int ct, int buf) {
    u16* dsth = bbuf + buf * (2 * BTILE);
    u16* dstl = dsth + BTILE;
    const u16* gh = wh + ct * 16 * 256;
    const u16* gl = wl + ct * 16 * 256;
    #pragma unroll
    for (int k = 0; k < 4; ++k) {
      int c = tid + k * 128;           // chunk 0..511
      int code = c >> 5;               // 0..15
      int ch8 = c & 31;                // 0..31 (8-channel chunks)
      int loff = code * BROW + ch8 * 8;
      int goff = code * 256 + ch8 * 8;
      async_copy_b128(dsth + loff, gh + goff);
      async_copy_b128(dstl + loff, gl + goff);
    }
  };

  float best[8];
  int bidx[8];
  #pragma unroll
  for (int j = 0; j < 8; ++j) { best[j] = 3.4e38f; bidx[j] = 0; }

  // Prologue: tile 0 in flight -> ready.
  stage_tile(0, 0);
  wait_async0();
  __syncthreads();

  for (int ct = 0; ct < 64; ++ct) {
    const int buf = ct & 1;
    if (ct < 63) stage_tile(ct + 1, buf ^ 1);   // overlaps with compute below

    const int code = ct * 16 + m;
    const float wsqv = wsq_lds[code];
    // B 32x16 bf16 layout: lane holds column n=m; low lanes K=k0+0..7 (e0..7),
    // k0+16..23 (e8..15); high lanes hold +8 within each 16-block.
    const u16* bh = bbuf + buf * (2 * BTILE) + m * BROW + hh * 8;
    const u16* bl = bh + BTILE;

    f32x8 acc;
    #pragma unroll
    for (int j = 0; j < 8; ++j) acc[j] = 0.0f;

    #pragma unroll
    for (int kc = 0; kc < 8; ++kc) {
      Frag Bh, Bl;
      Bh.q[0] = *(const uint4*)(bh + kc * 32);
      Bh.q[1] = *(const uint4*)(bh + kc * 32 + 16);
      Bl.q[0] = *(const uint4*)(bl + kc * 32);
      Bl.q[1] = *(const uint4*)(bl + kc * 32 + 16);
      acc = WMMA_BF16(false, Ah[kc], false, Bh.v, (short)0, acc, false, false);
      acc = WMMA_BF16(false, Al[kc], false, Bh.v, (short)0, acc, false, false);
      acc = WMMA_BF16(false, Ah[kc], false, Bl.v, (short)0, acc, false, false);
    }

    #pragma unroll
    for (int j = 0; j < 8; ++j) {
      float s = wsqv - 2.0f * acc[j];
      if (s < best[j]) { best[j] = s; bidx[j] = code; }  // strict <: lowest idx
    }

    if (ct < 63) {
      wait_async0();     // own portion of tile ct+1 arrived
      __syncthreads();   // all waves: tile ct+1 in LDS, compute(ct) finished
    }
  }

  // ---- Cross-lane argmin over the 16 columns per half-wave (wave32).
  #pragma unroll
  for (int msk = 1; msk <= 8; msk <<= 1) {
    #pragma unroll
    for (int j = 0; j < 8; ++j) {
      float ob = __shfl_xor(best[j], msk, 32);
      int oi = __shfl_xor(bidx[j], msk, 32);
      if (ob < best[j] || (ob == best[j] && oi < bidx[j])) {
        best[j] = ob;
        bidx[j] = oi;
      }
    }
  }
  if (m == 0) {
    // C layout: lanes 0-15 hold rows j (0..7); lanes 16-31 rows j+8.
    const int rbase = wv * 16 + hh * 8;
    #pragma unroll
    for (int j = 0; j < 8; ++j) idx_lds[rbase + j] = bidx[j];
  }
  __syncthreads();

  // ---- Phase 2a: gather winning fp32 codebook rows into LDS (coalesced).
  for (int i = tid; i < PIX * D; i += 128) {
    int p = i >> 8, ch = i & 255;
    wst[p * FSTRIDE + ch] = w[idx_lds[p] * 256 + ch];
  }
  __syncthreads();

  // ---- Phase 2b: coalesced NCHW store of quantized + loss partial.
  float lacc = 0.0f;
  for (int i = tid; i < PIX * D; i += 128) {
    int ch = i >> 6, p = i & 63;
    size_t go = xbase + (size_t)ch * 1024 + p;
    float wvv = wst[p * FSTRIDE + ch];
    float xv = x[go];                 // L2 hit (read in phase 1)
    out[1 + go] = wvv;
    float d = wvv - xv;
    lacc += d * d;
  }
  red[tid] = lacc;
  __syncthreads();
  for (int o = 64; o > 0; o >>= 1) {
    if (tid < o) red[tid] += red[tid + o];
    __syncthreads();
  }
  if (tid == 0) partial[wg] = red[0];   // deterministic (no float atomics)
}

// ---------------------------------------------------------------------------
// Kernel 2: reduce 1024 partials -> loss scalar. loss = 1.25 * mean(diff^2).
// ---------------------------------------------------------------------------
__global__ void vq_finish(const float* __restrict__ partial,
                          float* __restrict__ out) {
  __shared__ float s[256];
  int t = threadIdx.x;
  s[t] = partial[t] + partial[t + 256] + partial[t + 512] + partial[t + 768];
  __syncthreads();
  for (int o = 128; o > 0; o >>= 1) {
    if (t < o) s[t] += s[t + o];
    __syncthreads();
  }
  if (t == 0) out[0] = s[0] * (1.25f / 16777216.0f);
}

// ---------------------------------------------------------------------------
extern "C" void kernel_launch(void* const* d_in, const int* in_sizes, int n_in,
                              void* d_out, int out_size, void* d_ws,
                              size_t ws_size, hipStream_t stream) {
  const float* x = (const float*)d_in[0];   // [64,256,32,32] fp32
  const float* w = (const float*)d_in[1];   // [1024,256] fp32
  float* out = (float*)d_out;               // [1 + 64*256*32*32] fp32
  char* ws = (char*)d_ws;                   // needs ~1.06 MB
  u16* wh = (u16*)ws;                       // 512 KB  codebook hi
  u16* wl = (u16*)(ws + 524288);            // 512 KB  codebook lo
  float* wsq = (float*)(ws + 1048576);      // 4 KB    |w_k|^2
  float* partial = (float*)(ws + 1052672);  // 4 KB    per-WG loss partials

  vq_prep<<<1024, 256, 0, stream>>>(w, wh, wl, wsq);
  vq_main<<<1024, 128, 67584, stream>>>(x, w, wh, wl, wsq, out, partial);
  vq_finish<<<1, 256, 0, stream>>>(partial, out);
}
